// MuDeltaFormerLayer_57784490000788
// MI455X (gfx1250) — matrix operand order repaired and checked by reading, not compiled, not run on hardware
//
#include <hip/hip_runtime.h>

// ---------------------------------------------------------------------------
// MuDeltaFormer layer for MI455X (gfx1250, wave32, WMMA + TDM).
// All dense matrix work runs on v_wmma_f32_16x16x32_bf16; global->LDS tiles
// go through the Tensor Data Mover (tensor_load_to_lds) where available.
// ---------------------------------------------------------------------------

#define Bsz 4
#define Tsz 1024
#define Dsz 512
#define Hn  8
#define DK  64
#define DV  64
#define BH  (Bsz * Hn)

typedef __bf16 bhalf;
typedef bhalf v16bf __attribute__((ext_vector_type(16)));
typedef float v8f  __attribute__((ext_vector_type(8)));
typedef unsigned int u32x4 __attribute__((ext_vector_type(4)));
typedef int i32x4 __attribute__((ext_vector_type(4)));
typedef int i32x8 __attribute__((ext_vector_type(8)));

#define WMMA_BF16(a, b, c) \
  __builtin_amdgcn_wmma_f32_16x16x32_bf16(false, (a), false, (b), (short)0, (c), false, false)

#if defined(__has_builtin)
#if __has_builtin(__builtin_amdgcn_tensor_load_to_lds) && \
    __has_builtin(__builtin_amdgcn_s_wait_tensorcnt)
#define HAVE_TDM 1
#endif
#endif

#ifdef HAVE_TDM
// TDM 2-D tile load: bf16 tensor (rows x k_elems, row stride k_elems elems)
// -> LDS image with 64-elem rows padded to 68 (pad 2 DWORDs every 32 DWORDs).
// Issue from ONE wave only; EXEC is ignored by TDM.
__device__ inline void tdm_load_tile_bf16(unsigned lds_byte_addr,
                                          const bhalf* gsrc, unsigned k_elems,
                                          unsigned rows) {
  unsigned long long ga = (unsigned long long)(uintptr_t)gsrc;
  u32x4 g0;
  g0[0] = 1u;                                   // count=1, load, user mode
  g0[1] = lds_byte_addr;                        // LDS byte address
  g0[2] = (unsigned)(ga & 0xFFFFFFFFu);         // global_addr[31:0]
  g0[3] = (unsigned)((ga >> 32) & 0x1FFFFFFu) | (2u << 30);  // [56:32] | type=2
  i32x8 g1;
  // data_size=2B(code1) | pad_enable | pad_interval=32DW(code4) | pad_amount=2DW(code1)
  g1[0] = (int)((1u << 16) | (1u << 20) | (4u << 22) | (1u << 25));
  g1[1] = (int)((k_elems & 0xFFFFu) << 16);                 // tensor_dim0 lo16 @bit48
  g1[2] = (int)(((k_elems >> 16) & 0xFFFFu) |               // tensor_dim0 hi16
                ((rows & 0xFFFFu) << 16));                  // tensor_dim1 lo16
  g1[3] = (int)(((rows >> 16) & 0xFFFFu) | (64u << 16));    // dim1 hi16 | tile_dim0=64
  g1[4] = (int)64u;                                         // tile_dim1=64, tile_dim2=0
  g1[5] = (int)k_elems;                                     // tensor_dim0_stride lo32
  g1[6] = 0;
  g1[7] = 0;
  i32x4 z4 = {0, 0, 0, 0};
#if defined(__clang_major__) && (__clang_major__ >= 23)
  i32x8 z8 = {0, 0, 0, 0, 0, 0, 0, 0};
  __builtin_amdgcn_tensor_load_to_lds(g0, g1, z4, z4, z8, 0);
#else
  __builtin_amdgcn_tensor_load_to_lds(g0, g1, z4, z4, 0);
#endif
}
#endif

// ---- WMMA fragment loaders from LDS (layouts per CDNA5 ISA 7.12.2) ---------
// A 16x32 bf16 fragment: contiguous halves per lane (merges to ds_load_b64).
__device__ inline v16bf frag_a(const bhalf* base, int ld, int row0, int k0) {
  const int lane = threadIdx.x & 31;
  const bhalf* p = base + (row0 + (lane & 15)) * ld + k0 + ((lane >> 4) << 3);
  v16bf f;
#pragma unroll
  for (int e = 0; e < 8; ++e) f[e] = p[e];
#pragma unroll
  for (int e = 0; e < 8; ++e) f[8 + e] = p[16 + e];
  return f;
}
// B 32x16 from a matrix stored (N,K) row-major: B(k,n) = M[n][k]  (X * Y^T)
__device__ inline v16bf frag_bT(const bhalf* base, int ld, int n0, int k0) {
  const int lane = threadIdx.x & 31;
  const bhalf* p = base + (n0 + (lane & 15)) * ld + k0 + ((lane >> 4) << 3);
  v16bf f;
#pragma unroll
  for (int e = 0; e < 8; ++e) f[e] = p[e];
#pragma unroll
  for (int e = 0; e < 8; ++e) f[8 + e] = p[16 + e];
  return f;
}

// ---------------------------------------------------------------------------
__global__ void cvt_f32_to_bf16_kernel(const float* __restrict__ in,
                                       bhalf* __restrict__ out, int n) {
  for (int i = blockIdx.x * blockDim.x + threadIdx.x; i < n;
       i += gridDim.x * blockDim.x)
    out[i] = (bhalf)in[i];
}

// C[M,N] = A[M,K] * B[N,K]^T, bf16 inputs, f32 output. 64x64 tile / 4 waves.
__global__ __launch_bounds__(128) void gemm_tn_kernel(
    const bhalf* __restrict__ A, const bhalf* __restrict__ Bw,
    float* __restrict__ C, int M, int N, int K) {
  __shared__ __align__(128) bhalf As[64 * 68];
  __shared__ __align__(128) bhalf Bs[64 * 68];
  const int tid = threadIdx.x, wave = tid >> 5, lane = tid & 31;
  const int m0 = blockIdx.y * 64, n0 = blockIdx.x * 64;
  v8f acc[4] = {};
  for (int kk = 0; kk < K; kk += 64) {
#ifdef HAVE_TDM
    if (wave == 0) {
      tdm_load_tile_bf16((unsigned)(uintptr_t)&As[0],
                         A + (size_t)m0 * K + kk, (unsigned)(K - kk),
                         (unsigned)(M - m0));
      tdm_load_tile_bf16((unsigned)(uintptr_t)&Bs[0],
                         Bw + (size_t)n0 * K + kk, (unsigned)(K - kk),
                         (unsigned)(N - n0));
      __builtin_amdgcn_s_wait_tensorcnt((short)0);
    }
#else
    for (int idx = tid; idx < 64 * 64; idx += 128) {
      int r = idx >> 6, c = idx & 63;
      int gr = m0 + r, gk = kk + c, gn = n0 + r;
      As[r * 68 + c] = (gr < M && gk < K) ? A[(size_t)gr * K + gk] : (bhalf)0.f;
      Bs[r * 68 + c] = (gn < N && gk < K) ? Bw[(size_t)gn * K + gk] : (bhalf)0.f;
    }
#endif
    __syncthreads();
#pragma unroll
    for (int kc = 0; kc < 64; kc += 32) {
      v16bf af = frag_a(As, 68, wave * 16, kc);
#pragma unroll
      for (int t4 = 0; t4 < 4; ++t4) {
        v16bf bfr = frag_bT(Bs, 68, t4 * 16, kc);
        acc[t4] = WMMA_BF16(af, bfr, acc[t4]);
      }
    }
    __syncthreads();
  }
  const int cn = lane & 15;
  const int rb = m0 + wave * 16 + ((lane >> 4) << 3);
#pragma unroll
  for (int t4 = 0; t4 < 4; ++t4) {
    int gc = n0 + t4 * 16 + cn;
    if (gc < N) {
#pragma unroll
      for (int e = 0; e < 8; ++e) {
        int gr = rb + e;
        if (gr < M) C[(size_t)gr * N + gc] = acc[t4][e];
      }
    }
  }
}

// (b,t,h*64+e) f32  ->  [(b*8+h)][t][e] as f32 and/or bf16
__global__ void repack_heads_kernel(const float* __restrict__ in,
                                    float* __restrict__ outF,
                                    bhalf* __restrict__ outB, int total) {
  for (int idx = blockIdx.x * blockDim.x + threadIdx.x; idx < total;
       idx += gridDim.x * blockDim.x) {
    int e = idx & 63, hh = (idx >> 6) & 7, tt = (idx >> 9) & 1023, bb = idx >> 19;
    float v = in[idx];
    size_t o = ((size_t)(bb * 8 + hh) * 1024 + tt) * 64 + e;
    if (outF) outF[o] = v;
    if (outB) outB[o] = (bhalf)v;
  }
}

// p (b,t,16) -> sigmoid -> eta[bh][t], theta[bh][t]
__global__ void sigmoid_split_kernel(const float* __restrict__ pf,
                                     float* __restrict__ eta,
                                     float* __restrict__ theta) {
  int idx = blockIdx.x * blockDim.x + threadIdx.x;
  if (idx >= BH * Tsz) return;
  int t = idx & 1023, h = (idx >> 10) & 7, b = idx >> 13;
  size_t base = (size_t)(b * 1024 + t) * 16 + h * 2;
  eta[idx]   = 1.f / (1.f + __expf(-pf[base + 0]));
  theta[idx] = 1.f / (1.f + __expf(-pf[base + 1]));
}

// ---------------------------------------------------------------------------
// Flash-style masked-softmax stats + unit-lower-triangular block solve.
// One workgroup (4 waves) per (b,h). U_i = V_i - sum_{j<=i} A_ij U_j.
// ---------------------------------------------------------------------------
__global__ __launch_bounds__(128) void pass_solve_kernel(
    const bhalf* __restrict__ wB, const bhalf* __restrict__ kB,
    const float* __restrict__ vF, float* __restrict__ U,
    bhalf* __restrict__ Ub) {
  const int bh = blockIdx.x;
  const bhalf* wHead = wB + (size_t)bh * Tsz * DK;
  const bhalf* kHead = kB + (size_t)bh * Tsz * DK;
  const float* vHead = vF + (size_t)bh * Tsz * DV;
  float* uHead = U + (size_t)bh * Tsz * DV;
  bhalf* ubHead = Ub + (size_t)bh * Tsz * DV;

  __shared__ __align__(128) bhalf Wt[64 * 68];
  __shared__ __align__(128) bhalf Kt[64 * 68];
  __shared__ bhalf At[64 * 68];
  __shared__ bhalf UtT[64 * 68];  // transposed U_j tile: UtT[dv][s]
  __shared__ float Sc[64 * 66];   // score block; reused as U_fin block
  __shared__ float sm[64], sl[64], sinv[64];

  const int tid = threadIdx.x, wave = tid >> 5, lane = tid & 31;
  const int cn = lane & 15;
  const int rb = wave * 16 + ((lane >> 4) << 3);

#ifdef HAVE_TDM
#define LOAD_TILE(dst, src) \
  do { if (wave == 0) { \
    tdm_load_tile_bf16((unsigned)(uintptr_t)&dst[0], (src), DK, 64); \
    __builtin_amdgcn_s_wait_tensorcnt((short)0); } } while (0)
#else
#define LOAD_TILE(dst, src) \
  do { for (int idx = tid; idx < 64 * 64; idx += 128) { \
    int r = idx >> 6, c = idx & 63; \
    dst[r * 68 + c] = (src)[(size_t)r * DK + c]; } } while (0)
#endif

  for (int i = 0; i < 16; ++i) {
    LOAD_TILE(Wt, wHead + (size_t)i * 64 * DK);
    if (tid < 64) { sm[tid] = -1e30f; sl[tid] = 0.f; }
    __syncthreads();

    // ---- pass A: running row max / sumexp over column blocks j=0..i ----
    for (int j = 0; j <= i; ++j) {
      LOAD_TILE(Kt, kHead + (size_t)j * 64 * DK);
      __syncthreads();
      v8f sacc[4] = {};
#pragma unroll
      for (int kc = 0; kc < 64; kc += 32) {
        v16bf af = frag_a(Wt, 68, wave * 16, kc);
#pragma unroll
        for (int t4 = 0; t4 < 4; ++t4) {
          v16bf bfr = frag_bT(Kt, 68, t4 * 16, kc);
          sacc[t4] = WMMA_BF16(af, bfr, sacc[t4]);
        }
      }
#pragma unroll
      for (int t4 = 0; t4 < 4; ++t4)
#pragma unroll
        for (int e = 0; e < 8; ++e)
          Sc[(rb + e) * 66 + t4 * 16 + cn] = sacc[t4][e] * 0.125f;
      __syncthreads();
      if (tid < 64) {
        int lim = (j == i) ? tid : 64;  // strict mask on diagonal block
        if (lim > 0) {
          float m = sm[tid], l = sl[tid], bm = -1e30f;
          for (int c = 0; c < lim; ++c) bm = fmaxf(bm, Sc[tid * 66 + c]);
          float nm = fmaxf(m, bm), s = 0.f;
          for (int c = 0; c < lim; ++c) s += __expf(Sc[tid * 66 + c] - nm);
          sl[tid] = l * __expf(m - nm) + s;
          sm[tid] = nm;
        }
      }
      __syncthreads();
    }
    if (tid < 64) sinv[tid] = sl[tid] > 0.f ? 1.f / sl[tid] : 0.f;
    __syncthreads();

    // ---- pass B: accumulate sum_{j<i} A_ij @ U_j in registers ----
    v8f uacc[4] = {};
    for (int j = 0; j < i; ++j) {
      LOAD_TILE(Kt, kHead + (size_t)j * 64 * DK);
      for (int idx = tid; idx < 64 * 64; idx += 128) {  // transposed copy
        int r = idx >> 6, c = idx & 63;
        UtT[c * 68 + r] = ubHead[(size_t)(j * 64 + r) * DV + c];
      }
      __syncthreads();
      v8f sacc[4] = {};
#pragma unroll
      for (int kc = 0; kc < 64; kc += 32) {
        v16bf af = frag_a(Wt, 68, wave * 16, kc);
#pragma unroll
        for (int t4 = 0; t4 < 4; ++t4) {
          v16bf bfr = frag_bT(Kt, 68, t4 * 16, kc);
          sacc[t4] = WMMA_BF16(af, bfr, sacc[t4]);
        }
      }
#pragma unroll
      for (int t4 = 0; t4 < 4; ++t4)
#pragma unroll
        for (int e = 0; e < 8; ++e)
          Sc[(rb + e) * 66 + t4 * 16 + cn] = sacc[t4][e] * 0.125f;
      __syncthreads();
      for (int idx = tid; idx < 64 * 64; idx += 128) {
        int r = idx >> 6, c = idx & 63;
        At[r * 68 + c] = (bhalf)(__expf(Sc[r * 66 + c] - sm[r]) * sinv[r]);
      }
      __syncthreads();
#pragma unroll
      for (int kc = 0; kc < 64; kc += 32) {
        v16bf af = frag_a(At, 68, wave * 16, kc);
#pragma unroll
        for (int t4 = 0; t4 < 4; ++t4) {
          v16bf bfr = frag_bT(UtT, 68, t4 * 16, kc);  // B(k=s,n=dv)=UtT[dv][s]
          uacc[t4] = WMMA_BF16(af, bfr, uacc[t4]);
        }
      }
      __syncthreads();
    }

    // ---- diagonal block: strict-lower A_ii, then sequential solve ----
    LOAD_TILE(Kt, kHead + (size_t)i * 64 * DK);
    __syncthreads();
    {
      v8f sacc[4] = {};
#pragma unroll
      for (int kc = 0; kc < 64; kc += 32) {
        v16bf af = frag_a(Wt, 68, wave * 16, kc);
#pragma unroll
        for (int t4 = 0; t4 < 4; ++t4) {
          v16bf bfr = frag_bT(Kt, 68, t4 * 16, kc);
          sacc[t4] = WMMA_BF16(af, bfr, sacc[t4]);
        }
      }
#pragma unroll
      for (int t4 = 0; t4 < 4; ++t4)
#pragma unroll
        for (int e = 0; e < 8; ++e)
          Sc[(rb + e) * 66 + t4 * 16 + cn] = sacc[t4][e] * 0.125f;
    }
    __syncthreads();
    for (int idx = tid; idx < 64 * 64; idx += 128) {
      int r = idx >> 6, c = idx & 63;
      At[r * 68 + c] =
          (c < r) ? (bhalf)(__expf(Sc[r * 66 + c] - sm[r]) * sinv[r]) : (bhalf)0.f;
    }
    __syncthreads();
    // U_fin = V_i - uacc (reuse Sc)
#pragma unroll
    for (int t4 = 0; t4 < 4; ++t4)
#pragma unroll
      for (int e = 0; e < 8; ++e)
        Sc[(rb + e) * 66 + t4 * 16 + cn] = -uacc[t4][e];
    __syncthreads();
    for (int idx = tid; idx < 64 * 64; idx += 128) {
      int r = idx >> 6, c = idx & 63;
      Sc[r * 66 + c] += vHead[(size_t)(i * 64 + r) * DV + c];
    }
    __syncthreads();
    if (tid < 64) {  // one thread per dv column; column-private, no barriers
      int c = tid;
      for (int r = 1; r < 64; ++r) {
        float a = Sc[r * 66 + c];
        for (int s = 0; s < r; ++s) a -= (float)At[r * 68 + s] * Sc[s * 66 + c];
        Sc[r * 66 + c] = a;
      }
    }
    __syncthreads();
    for (int idx = tid; idx < 64 * 64; idx += 128) {
      int r = idx >> 6, c = idx & 63;
      float uv = Sc[r * 66 + c];
      uHead[(size_t)(i * 64 + r) * DV + c] = uv;
      ubHead[(size_t)(i * 64 + r) * DV + c] = (bhalf)uv;
    }
    __syncthreads();
  }
#undef LOAD_TILE
}

// ---------------------------------------------------------------------------
// Sequential scan: one workgroup (8 waves, 256 threads) per (b,h).
// M and S live in VGPRs (16 f32 each per thread); X/A/Bm live in LDS with
// transposed bf16 mirrors so every WMMA B-fragment is a contiguous LDS read.
// ---------------------------------------------------------------------------
__global__ __launch_bounds__(256) void scan_kernel(
    const float* __restrict__ U, const float* __restrict__ kh,
    const float* __restrict__ qh, const float* __restrict__ eta,
    const float* __restrict__ theta, const float* __restrict__ S0,
    const float* __restrict__ M0, bhalf* __restrict__ O) {
  const int bh = blockIdx.x, b = bh >> 3, h = bh & 7;
  const float* uHead = U + (size_t)bh * Tsz * DV;
  const float* kHead = kh + (size_t)bh * Tsz * DK;
  const float* qHead = qh + (size_t)bh * Tsz * DK;
  const float* etaH = eta + (size_t)bh * Tsz;
  const float* thH = theta + (size_t)bh * Tsz;

  __shared__ float Xf[64 * 66];
  __shared__ bhalf Xb[64 * 68];   // X row-major
  __shared__ bhalf Xbt[64 * 68];  // X transposed (for B(k,n)=X[k][n] frags)
  __shared__ bhalf Ab[64 * 68];   // A row-major
  __shared__ bhalf Abt[64 * 68];  // A transposed
  __shared__ bhalf Bmb[64 * 68];
  __shared__ float red[256];
  __shared__ float su[64], sk[64], sq[64];

  const int tid = threadIdx.x, wave = tid >> 5, lane = tid & 31;
  const int row = tid >> 2, c0 = (tid & 3) * 16;  // M/S register ownership
  const int wrow = (wave & 3) * 16, wcol = (wave >> 2) * 32;  // WMMA tiles
  const int cn = lane & 15;
  const int rb = wrow + ((lane >> 4) << 3);

  float Sreg[16], Mreg[16];
#pragma unroll
  for (int jj = 0; jj < 16; ++jj) {
    Sreg[jj] = S0[((size_t)h * 64 + row) * 64 + c0 + jj];
    Mreg[jj] = M0[((size_t)h * 64 + row) * 64 + c0 + jj];
  }

  const float nsA[6] = {3955.f / 1024, 3735.f / 1024, 3799.f / 1024,
                        4019.f / 1024, 2677.f / 1024, 2172.f / 1024};
  const float nsB[6] = {-8306.f / 1024, -6681.f / 1024, -6499.f / 1024,
                        -6385.f / 1024, -3029.f / 1024, -1833.f / 1024};
  const float nsC[6] = {5008.f / 1024, 3463.f / 1024, 3211.f / 1024,
                        2906.f / 1024, 1162.f / 1024, 682.f / 1024};

  for (int t = 0; t < Tsz; ++t) {
    if (tid < 64) {
      su[tid] = uHead[(size_t)t * DV + tid];
      sk[tid] = kHead[(size_t)t * DK + tid];
      sq[tid] = qHead[(size_t)t * DK + tid];
    }
    const float et = etaH[t], th = thH[t];
    __syncthreads();

    // M = th*M + (1-th) * u k^T ; Frobenius norm
    float ss = 0.f;
    const float uu = su[row];
#pragma unroll
    for (int jj = 0; jj < 16; ++jj) {
      Mreg[jj] = th * Mreg[jj] + (1.f - th) * uu * sk[c0 + jj];
      ss += Mreg[jj] * Mreg[jj];
    }
    red[tid] = ss;
    __syncthreads();
    for (int off = 128; off > 0; off >>= 1) {
      if (tid < off) red[tid] += red[tid + off];
      __syncthreads();
    }
    const float inv = 1.f / (sqrtf(red[0]) + 1e-7f);
    __syncthreads();
#pragma unroll
    for (int jj = 0; jj < 16; ++jj) {
      float xv = Mreg[jj] * inv;
      bhalf xb = (bhalf)xv;
      Xf[row * 66 + c0 + jj] = xv;
      Xb[row * 68 + c0 + jj] = xb;
      Xbt[(c0 + jj) * 68 + row] = xb;
    }
    __syncthreads();

    // ---- 6 Newton-Schulz iterations, all matmuls on WMMA ----
#pragma unroll
    for (int it = 0; it < 6; ++it) {
      // A = X X^T
      v8f a0 = {}, a1 = {};
#pragma unroll
      for (int kc = 0; kc < 64; kc += 32) {
        v16bf af = frag_a(Xb, 68, wrow, kc);
        v16bf b0 = frag_bT(Xb, 68, wcol, kc);
        v16bf b1 = frag_bT(Xb, 68, wcol + 16, kc);
        a0 = WMMA_BF16(af, b0, a0);
        a1 = WMMA_BF16(af, b1, a1);
      }
      __syncthreads();
#pragma unroll
      for (int e = 0; e < 8; ++e) {
        bhalf v0 = (bhalf)a0[e], v1 = (bhalf)a1[e];
        Ab[(rb + e) * 68 + wcol + cn] = v0;
        Ab[(rb + e) * 68 + wcol + 16 + cn] = v1;
        Abt[(wcol + cn) * 68 + rb + e] = v0;
        Abt[(wcol + 16 + cn) * 68 + rb + e] = v1;
      }
      __syncthreads();
      // A2 = A A ; Bm = b*A + c*A2 (f32 A term straight from registers)
      v8f q0 = {}, q1 = {};
#pragma unroll
      for (int kc = 0; kc < 64; kc += 32) {
        v16bf af = frag_a(Ab, 68, wrow, kc);
        v16bf b0 = frag_bT(Abt, 68, wcol, kc);       // B(k,n)=A[k][n]
        v16bf b1 = frag_bT(Abt, 68, wcol + 16, kc);
        q0 = WMMA_BF16(af, b0, q0);
        q1 = WMMA_BF16(af, b1, q1);
      }
#pragma unroll
      for (int e = 0; e < 8; ++e) {
        Bmb[(rb + e) * 68 + wcol + cn] = (bhalf)(nsB[it] * a0[e] + nsC[it] * q0[e]);
        Bmb[(rb + e) * 68 + wcol + 16 + cn] =
            (bhalf)(nsB[it] * a1[e] + nsC[it] * q1[e]);
      }
      __syncthreads();
      // X = a*X + Bm @ X
      v8f x0 = {}, x1 = {};
#pragma unroll
      for (int kc = 0; kc < 64; kc += 32) {
        v16bf af = frag_a(Bmb, 68, wrow, kc);
        v16bf b0 = frag_bT(Xbt, 68, wcol, kc);       // B(k,n)=X[k][n]
        v16bf b1 = frag_bT(Xbt, 68, wcol + 16, kc);
        x0 = WMMA_BF16(af, b0, x0);
        x1 = WMMA_BF16(af, b1, x1);
      }
      __syncthreads();  // all Xb/Xbt reads complete before overwrite
#pragma unroll
      for (int e = 0; e < 8; ++e) {
        float xv0 = nsA[it] * Xf[(rb + e) * 66 + wcol + cn] + x0[e];
        float xv1 = nsA[it] * Xf[(rb + e) * 66 + wcol + 16 + cn] + x1[e];
        bhalf xb0 = (bhalf)xv0, xb1 = (bhalf)xv1;
        Xf[(rb + e) * 66 + wcol + cn] = xv0;
        Xf[(rb + e) * 66 + wcol + 16 + cn] = xv1;
        Xb[(rb + e) * 68 + wcol + cn] = xb0;
        Xb[(rb + e) * 68 + wcol + 16 + cn] = xb1;
        Xbt[(wcol + cn) * 68 + rb + e] = xb0;
        Xbt[(wcol + 16 + cn) * 68 + rb + e] = xb1;
      }
      __syncthreads();
    }

    // S += eta * X ; o = S q (per-thread partial dot + 4-way reduce)
    float part = 0.f;
#pragma unroll
    for (int jj = 0; jj < 16; ++jj) {
      Sreg[jj] += et * Xf[row * 66 + c0 + jj];
      part += Sreg[jj] * sq[c0 + jj];
    }
    red[tid] = part;
    __syncthreads();
    if (tid < 64) {
      float o = red[tid * 4] + red[tid * 4 + 1] + red[tid * 4 + 2] + red[tid * 4 + 3];
      O[(size_t)(b * Tsz + t) * (Hn * DV) + h * DV + tid] = (bhalf)o;
    }
    __syncthreads();
  }
}

// ---------------------------------------------------------------------------
extern "C" void kernel_launch(void* const* d_in, const int* in_sizes, int n_in,
                              void* d_out, int out_size, void* d_ws,
                              size_t ws_size, hipStream_t stream) {
  (void)in_sizes; (void)n_in; (void)out_size; (void)ws_size;
  const float* x  = (const float*)d_in[0];
  const float* Wq = (const float*)d_in[1];
  const float* Wk = (const float*)d_in[2];
  const float* Wv = (const float*)d_in[3];
  const float* Ww = (const float*)d_in[4];
  const float* Wp = (const float*)d_in[5];
  const float* Wo = (const float*)d_in[6];
  const float* S0 = (const float*)d_in[7];
  const float* M0 = (const float*)d_in[8];
  float* out = (float*)d_out;

  char* ws = (char*)d_ws;
  size_t off = 0;
  auto alloc = [&](size_t bytes) -> void* {
    void* p = ws + off;
    off = (off + bytes + 255) & ~(size_t)255;
    return p;
  };

  const int NX = Bsz * Tsz * Dsz;          // 2,097,152
  const int NW = Dsz * Dsz;                // 262,144
  const int NH = BH * Tsz * 64;            // 2,097,152

  bhalf* xb  = (bhalf*)alloc((size_t)NX * 2);
  bhalf* Wqb = (bhalf*)alloc((size_t)NW * 2);
  bhalf* Wkb = (bhalf*)alloc((size_t)NW * 2);
  bhalf* Wvb = (bhalf*)alloc((size_t)NW * 2);
  bhalf* Wwb = (bhalf*)alloc((size_t)NW * 2);
  bhalf* Wpb = (bhalf*)alloc((size_t)16 * Dsz * 2);
  bhalf* Wob = (bhalf*)alloc((size_t)NW * 2);
  float* qf  = (float*)alloc((size_t)NX * 4);
  float* kf  = (float*)alloc((size_t)NX * 4);
  float* vf  = (float*)alloc((size_t)NX * 4);
  float* wf  = (float*)alloc((size_t)NX * 4);
  float* pf  = (float*)alloc((size_t)Bsz * Tsz * 16 * 4);
  float* qhF = (float*)alloc((size_t)NH * 4);
  float* khF = (float*)alloc((size_t)NH * 4);
  float* vhF = (float*)alloc((size_t)NH * 4);
  bhalf* khB = (bhalf*)alloc((size_t)NH * 2);
  bhalf* whB = (bhalf*)alloc((size_t)NH * 2);
  float* etaB = (float*)alloc((size_t)BH * Tsz * 4);
  float* thB  = (float*)alloc((size_t)BH * Tsz * 4);
  float* Uf  = (float*)alloc((size_t)NH * 4);
  bhalf* Ubb = (bhalf*)alloc((size_t)NH * 2);
  bhalf* Ob  = (bhalf*)alloc((size_t)NX * 2);

  auto cvt = [&](const float* src, bhalf* dst, int n) {
    cvt_f32_to_bf16_kernel<<<dim3((n + 255) / 256), dim3(256), 0, stream>>>(src, dst, n);
  };
  cvt(x, xb, NX);
  cvt(Wq, Wqb, NW); cvt(Wk, Wkb, NW); cvt(Wv, Wvb, NW); cvt(Ww, Wwb, NW);
  cvt(Wp, Wpb, 16 * Dsz); cvt(Wo, Wob, NW);

  // projections (WMMA GEMM, C = x * W^T)
  dim3 gbig(Dsz / 64, (Bsz * Tsz) / 64);
  gemm_tn_kernel<<<gbig, 128, 0, stream>>>(xb, Wqb, qf, Bsz * Tsz, Dsz, Dsz);
  gemm_tn_kernel<<<gbig, 128, 0, stream>>>(xb, Wkb, kf, Bsz * Tsz, Dsz, Dsz);
  gemm_tn_kernel<<<gbig, 128, 0, stream>>>(xb, Wvb, vf, Bsz * Tsz, Dsz, Dsz);
  gemm_tn_kernel<<<gbig, 128, 0, stream>>>(xb, Wwb, wf, Bsz * Tsz, Dsz, Dsz);
  gemm_tn_kernel<<<dim3(1, (Bsz * Tsz) / 64), 128, 0, stream>>>(
      xb, Wpb, pf, Bsz * Tsz, 16, Dsz);

  // head-major repacks
  repack_heads_kernel<<<dim3(2048), dim3(256), 0, stream>>>(qf, qhF, nullptr, NH);
  repack_heads_kernel<<<dim3(2048), dim3(256), 0, stream>>>(kf, khF, khB, NH);
  repack_heads_kernel<<<dim3(2048), dim3(256), 0, stream>>>(vf, vhF, nullptr, NH);
  repack_heads_kernel<<<dim3(2048), dim3(256), 0, stream>>>(wf, nullptr, whB, NH);
  sigmoid_split_kernel<<<dim3((BH * Tsz + 255) / 256), dim3(256), 0, stream>>>(
      pf, etaB, thB);

  // flash-softmax stats + triangular solve -> U
  pass_solve_kernel<<<dim3(BH), 128, 0, stream>>>(whB, khB, vhF, Uf, Ubb);

  // sequential momentum + Newton-Schulz scan -> O
  scan_kernel<<<dim3(BH), 256, 0, stream>>>(Uf, khF, qhF, etaB, thB, S0, M0, Ob);

  // final projection: out = O @ Wo^T
  gemm_tn_kernel<<<gbig, 128, 0, stream>>>(Ob, Wob, out, Bsz * Tsz, Dsz, Dsz);
}
